// GraphConvolution_11579231830749
// MI455X (gfx1250) — matrix operand order repaired and checked by reading, not compiled, not run on hardware
//
#include <hip/hip_runtime.h>

typedef __attribute__((ext_vector_type(16))) __bf16 v16bf;
typedef __attribute__((ext_vector_type(8)))  float  v8f;

#define N_NODES 100000
#define N_EDGES 3200000
#define D_IN    512
#define D_OUT   256

// ---------------------------------------------------------------------------
// Workspace layout (bytes):
//   [0, 102400000)            pre_sup  (N_NODES x D_OUT fp32)
//   [102400000, +262144)      W_hi fragments (131072 bf16, WMMA-B swizzled)
//   [.., +262144)             W_lo fragments
//   [.., +1024)               colnorm2 (256 fp32)
// ---------------------------------------------------------------------------
#define WS_PRESUP_BYTES (102400000L)
#define W_ELEMS (D_IN * D_OUT)

// Swizzle W (fp32 row-major [512,256]) into per-lane-contiguous bf16 hi/lo
// fragments for the V_WMMA 16-bit B-matrix (32x16, K x N), in contiguous k order:
//   lane = g*16 + (n&15), element e (0..15): K = kstep*32 + g*16 + e.
__global__ void swizzle_W_kernel(const float* __restrict__ W,
                                 __bf16* __restrict__ whi,
                                 __bf16* __restrict__ wlo) {
    int t = blockIdx.x * blockDim.x + threadIdx.x;
    if (t >= W_ELEMS) return;
    int k = t >> 8;          // row of W (K index)
    int n = t & 255;         // col of W (N index)
    float f  = W[t];
    __bf16 hi = (__bf16)f;
    __bf16 lo = (__bf16)(f - (float)hi);
    int kstep = k >> 5;
    int kin   = k & 31;
    int g     = kin >> 4;
    int e     = kin & 15;
    int ntile = n >> 4;
    int lane  = g * 16 + (n & 15);
    int idx   = (((ntile * 16 + kstep) * 32) + lane) * 16 + e;
    whi[idx] = hi;
    wlo[idx] = lo;
}

// Seed d_out with the bias (agg starts at b) and zero the column-norm scratch.
__global__ void init_out_kernel(float* __restrict__ out,
                                const float* __restrict__ b,
                                float* __restrict__ colnorm2) {
    long total = (long)N_NODES * D_OUT;
    long i = (long)blockIdx.x * blockDim.x + threadIdx.x;
    if (i < D_OUT) colnorm2[i] = 0.0f;
    long stride = (long)gridDim.x * blockDim.x;
    for (; i < total; i += stride)
        out[i] = b[i & (D_OUT - 1)];
}

// Dense GEMM pre_sup = x @ W via split-bf16 WMMA (hi*hi + lo*hi + hi*lo).
// Block: 256 threads (8 waves) covering 16 rows x all 256 cols.
// The block's 16x512 fp32 A-tile is loaded coalesced ONCE, split into
// bf16 hi/lo, and stored to LDS pre-swizzled into a contiguous-k (k = 16*half + e) 16x32 WMMA
// A-fragment layout. Inner loop = ds_load_b128 (A) + global_load_b128 (B)
// + 6x v_wmma per K-step; no per-iteration conversion VALU.
__global__ __launch_bounds__(256) void gemm_wmma_kernel(
        const float* __restrict__ x,
        const __bf16* __restrict__ whi,
        const __bf16* __restrict__ wlo,
        float* __restrict__ presup) {
    // [ks][lane][e] fragments: 16 * 32 * 16 bf16 = 16 KB each
    __shared__ __align__(32) __bf16 lds_hi[16 * 32 * 16];
    __shared__ __align__(32) __bf16 lds_lo[16 * 32 * 16];

    const int t    = threadIdx.x;
    const int wave = t >> 5;
    const int lane = t & 31;
    const int m0   = blockIdx.x * 16;
    const int r    = lane & 15;
    const int g    = lane >> 4;
    const int n0   = wave * 32;
    const int nt0  = wave * 2;

    // ---- Stage A tile: 16 rows x 512 K fp32 -> split-bf16 fragments in LDS.
    // Flat element f = row*512 + k; 256 threads x float4 x 8 iters, coalesced.
#pragma unroll
    for (int i = 0; i < 8; ++i) {
        int f   = i * 1024 + t * 4;
        int row = f >> 9;
        int k0  = f & 511;
        float4 v = *(const float4*)(x + (long)(m0 + row) * D_IN + k0);
        float vv[4] = {v.x, v.y, v.z, v.w};
#pragma unroll
        for (int j = 0; j < 4; ++j) {
            int k   = k0 + j;
            int ks  = k >> 5;
            int kin = k & 31;
            int gg  = kin >> 4;                    // which lane-half holds it
            int e   = kin & 15;                    // fragment element index
            int idx = ((ks * 32) + gg * 16 + row) * 16 + e;
            __bf16 h = (__bf16)vv[j];
            lds_hi[idx] = h;
            lds_lo[idx] = (__bf16)(vv[j] - (float)h);
        }
    }
    __syncthreads();

    v8f acc0 = {};
    v8f acc1 = {};

    for (int ks = 0; ks < D_IN / 32; ++ks) {
        // A fragments: one 32B LDS read per matrix per lane
        const v16bf a_hi = *(const v16bf*)(lds_hi + ((ks * 32) + lane) * 16);
        const v16bf a_lo = *(const v16bf*)(lds_lo + ((ks * 32) + lane) * 16);

        // B fragments: contiguous 32B per lane (pre-swizzled, L2-resident)
        const long fb0 = (long)(((nt0    ) * 16 + ks) * 32 + lane) * 16;
        const long fb1 = (long)(((nt0 + 1) * 16 + ks) * 32 + lane) * 16;
        v16bf bh0 = *(const v16bf*)(whi + fb0);
        v16bf bl0 = *(const v16bf*)(wlo + fb0);
        v16bf bh1 = *(const v16bf*)(whi + fb1);
        v16bf bl1 = *(const v16bf*)(wlo + fb1);

        // split-bf16 accumulation (drop lo*lo; error ~2^-16 relative)
        acc0 = __builtin_amdgcn_wmma_f32_16x16x32_bf16(false, a_hi, false, bh0,
                                                       (short)0, acc0, false, false);
        acc0 = __builtin_amdgcn_wmma_f32_16x16x32_bf16(false, a_lo, false, bh0,
                                                       (short)0, acc0, false, false);
        acc0 = __builtin_amdgcn_wmma_f32_16x16x32_bf16(false, a_hi, false, bl0,
                                                       (short)0, acc0, false, false);
        acc1 = __builtin_amdgcn_wmma_f32_16x16x32_bf16(false, a_hi, false, bh1,
                                                       (short)0, acc1, false, false);
        acc1 = __builtin_amdgcn_wmma_f32_16x16x32_bf16(false, a_lo, false, bh1,
                                                       (short)0, acc1, false, false);
        acc1 = __builtin_amdgcn_wmma_f32_16x16x32_bf16(false, a_hi, false, bl1,
                                                       (short)0, acc1, false, false);
    }

    // C/D layout: VGPR v, lanes 0-15 -> M=v, N=lane; lanes 16-31 -> M=v+8
    float* pd = presup + n0 + r;
#pragma unroll
    for (int v = 0; v < 8; ++v) {
        long m = m0 + v + 8 * g;
        pd[m * D_OUT]      = acc0[v];
        pd[m * D_OUT + 16] = acc1[v];
    }
}

// Edge-parallel SpMM: out[row] += val * pre_sup[col].
// Thread t handles feature column t -> coalesced gather + coalesced atomics.
__global__ __launch_bounds__(256) void spmm_edges_kernel(
        const float* __restrict__ presup,
        const int*   __restrict__ erow,
        const int*   __restrict__ ecol,
        const float* __restrict__ evals,
        float* __restrict__ out) {
    const int t  = threadIdx.x;
    const int e0 = blockIdx.x * 16;
#pragma unroll 4
    for (int i = 0; i < 16; ++i) {
        int   e = e0 + i;
        int   r = erow[e];
        int   c = ecol[e];
        float v = evals[e];
        float gv = presup[(long)c * D_OUT + t] * v;
        atomicAdd(&out[(long)r * D_OUT + t], gv);
    }
}

// Per-column sum of squares (thread t = column t -> coalesced), one atomic.
__global__ __launch_bounds__(256) void colnorm_kernel(
        const float* __restrict__ out, float* __restrict__ colnorm2) {
    const int t  = threadIdx.x;
    const long r0 = (long)blockIdx.x * 400;
    float s = 0.0f;
    for (int i = 0; i < 400; ++i) {
        float v = out[(r0 + i) * D_OUT + t];
        s += v * v;
    }
    atomicAdd(&colnorm2[t], s);
}

// out = softplus(out / max(||col||_2, 1e-12)), numerically stable.
__global__ void finalize_kernel(float* __restrict__ out,
                                const float* __restrict__ colnorm2) {
    long total = (long)N_NODES * D_OUT;
    long i = (long)blockIdx.x * blockDim.x + threadIdx.x;
    long stride = (long)gridDim.x * blockDim.x;
    for (; i < total; i += stride) {
        float nrm = sqrtf(colnorm2[i & (D_OUT - 1)]);
        float d   = fmaxf(nrm, 1e-12f);
        float xv  = out[i] / d;
        out[i] = fmaxf(xv, 0.0f) + log1pf(expf(-fabsf(xv)));
    }
}

extern "C" void kernel_launch(void* const* d_in, const int* in_sizes, int n_in,
                              void* d_out, int out_size, void* d_ws, size_t ws_size,
                              hipStream_t stream) {
    const float* x     = (const float*)d_in[0];
    const int*   erow  = (const int*)  d_in[1];
    const int*   ecol  = (const int*)  d_in[2];
    const float* evals = (const float*)d_in[3];
    const float* W     = (const float*)d_in[4];
    const float* b     = (const float*)d_in[5];
    float* out = (float*)d_out;

    char*   ws     = (char*)d_ws;
    float*  presup = (float*)ws;
    __bf16* whi    = (__bf16*)(ws + WS_PRESUP_BYTES);
    __bf16* wlo    = whi + W_ELEMS;
    float*  cn2    = (float*)(wlo + W_ELEMS);

    swizzle_W_kernel<<<(W_ELEMS + 255) / 256, 256, 0, stream>>>(W, whi, wlo);
    init_out_kernel<<<4096, 256, 0, stream>>>(out, b, cn2);
    gemm_wmma_kernel<<<N_NODES / 16, 256, 0, stream>>>(x, whi, wlo, presup);
    spmm_edges_kernel<<<N_EDGES / 16, 256, 0, stream>>>(presup, erow, ecol, evals, out);
    colnorm_kernel<<<N_NODES / 400, 256, 0, stream>>>(out, cn2);
    finalize_kernel<<<4096, 256, 0, stream>>>(out, cn2);
}
